// PointerNetwork_54649163874573
// MI455X (gfx1250) — matrix operand order repaired and checked by reading, not compiled, not run on hardware
//
#include <hip/hip_runtime.h>
#include <math.h>

#define BB 32
#define SS 64
#define DD 8
#define HH 256
#define G4 1024   // 4*H
#define KC 512    // K-concat: [x | h]
#define NWG 16
#define NTHR 256

typedef __attribute__((ext_vector_type(16))) _Float16 v16h;
typedef __attribute__((ext_vector_type(8)))  _Float16 v8h;
typedef __attribute__((ext_vector_type(8)))  float    v8f;

// ---- CDNA5 async global->LDS copy (ASYNCcnt path), with safe fallback ----
#if defined(__has_builtin)
#  if __has_builtin(__builtin_amdgcn_global_load_async_to_lds_b128)
#    define USE_ASYNC_LDS 1
#  endif
#endif
#ifndef USE_ASYNC_LDS
#  define USE_ASYNC_LDS 0
#endif

#if USE_ASYNC_LDS
// Builtin signature (from hipcc diagnostic): arg0 is a pointer to a 16-byte
// int vector in the global address space (printed as "v4i __device__ *").
typedef int v4i __attribute__((vector_size(4 * sizeof(int))));
typedef __attribute__((address_space(1))) v4i as1_v4i;
typedef __attribute__((address_space(3))) v4i as3_v4i;
__device__ __forceinline__ void async_b128(_Float16* dst_lds, const _Float16* src_glob) {
  // AS3 pointers are 32-bit LDS offsets (low 32 bits of the flat address);
  // AS1 is the plain 64-bit global address. Route casts through integers so
  // they always compile.
  __builtin_amdgcn_global_load_async_to_lds_b128(
      (as1_v4i*)(unsigned long long)(const void*)src_glob,
      (as3_v4i*)(unsigned)(unsigned long long)(void*)dst_lds,
      0, 0);
}
__device__ __forceinline__ void wait_async0() {
#if defined(__has_builtin) && __has_builtin(__builtin_amdgcn_s_wait_asynccnt)
  __builtin_amdgcn_s_wait_asynccnt(0);
#else
  asm volatile("s_wait_asynccnt 0x0" ::: "memory");
#endif
}
#endif

struct Ctx {
  const float* inputs;
  const float* W_emb;
  const float* b_emb;
  const float* bias0;
  const float* bias1;
  const float* bias2;
  const float* bias3;
  const float* vvec;
  const _Float16* Wp;      // [4][1024][512] packed fp16 LSTM weights
  const _Float16* W1h;     // [256][256]
  const _Float16* W2h;     // [256][256]
  _Float16* emb;           // [B][S][H]
  _Float16* h0g0; _Float16* h0g1;   // encoder layer0 h ping-pong [B][H]
  _Float16* h1g0; _Float16* h1g1;   // encoder layer1 h ping-pong
  _Float16* hd00; _Float16* hd01;   // decoder layer0 h ping-pong
  _Float16* hd10; _Float16* hd11;   // decoder layer1 h ping-pong
  _Float16* enc_out;       // [B][S][H]
  _Float16* dec_in;        // [B][H]
  float* flag;             // [B][S] (mask == cur[...,-1])
  unsigned* bar;           // [0]=count, [1]=generation
  float* out_ptr;          // [B][S][S]
  float* out_idx;          // [B][S]
};

__device__ __forceinline__ float sigf(float x) { return 1.0f / (1.0f + __expf(-x)); }

__device__ __forceinline__ v8f zero8f() {
  v8f z;
#pragma unroll
  for (int i = 0; i < 8; ++i) z[i] = 0.0f;
  return z;
}
__device__ __forceinline__ v8h zero8h() {
  v8h z;
#pragma unroll
  for (int i = 0; i < 8; ++i) z[i] = (_Float16)0.0f;
  return z;
}
__device__ __forceinline__ v16h frag_cat(v8h lo, v8h hi) {
  v16h f;
#pragma unroll
  for (int i = 0; i < 8; ++i) { f[i] = lo[i]; f[8 + i] = hi[i]; }
  return f;
}
// A fragment of a 16x32 fp16 tile, row-major, ld in halfs.
// Lanes 0-15: m=lane, K 0-7 & 16-23; lanes 16-31: m=lane-16, K 8-15 & 24-31.
__device__ __forceinline__ v16h load_a(const _Float16* tile, int ld, int lane) {
  int m  = lane & 15;
  int kb = (lane < 16) ? 0 : 8;
  const _Float16* p = tile + m * ld + kb;
  return frag_cat(*(const v8h*)p, *(const v8h*)(p + 16));
}
// B fragment of a 32x16 tile; source is W row-major [N][K] (B(k,n)=W[n][k]).
// Lanes 0-15: n=lane, K 0-7 & 16-23; lanes 16-31: n=lane-16, K 8-15 & 24-31.
__device__ __forceinline__ v16h load_b(const _Float16* tile, int ld, int lane) {
  int n  = lane & 15;
  int kb = (lane < 16) ? 0 : 8;
  const _Float16* p = tile + n * ld + kb;
  return frag_cat(*(const v8h*)p, *(const v8h*)(p + 16));
}

// Device-wide generation barrier across the 16 persistent workgroups.
__device__ void gbar(unsigned* bar, unsigned& lgen) {
  __syncthreads();
  ++lgen;
  if (threadIdx.x == 0) {
    __threadfence();
    unsigned a = __hip_atomic_fetch_add(&bar[0], 1u, __ATOMIC_ACQ_REL, __HIP_MEMORY_SCOPE_AGENT);
    if (a == NWG - 1) {
      __hip_atomic_store(&bar[0], 0u, __ATOMIC_RELAXED, __HIP_MEMORY_SCOPE_AGENT);
      __hip_atomic_store(&bar[1], lgen, __ATOMIC_RELEASE, __HIP_MEMORY_SCOPE_AGENT);
    } else {
      while (__hip_atomic_load(&bar[1], __ATOMIC_ACQUIRE, __HIP_MEMORY_SCOPE_AGENT) < lgen) {
        __builtin_amdgcn_s_sleep(2);
      }
    }
    __threadfence();
  }
  __syncthreads();
}

// One LSTM cell step for this WGP's 16 hidden units (all 4 gates).
// A = [x | h] (32x512) staged in LDS; B slices are LDS-resident weights.
__device__ void lstm_cell(const Ctx& c, const _Float16* wlds, _Float16* abuf, float* gbufS,
                          float* c_arr, int lw, const _Float16* xsrc, int xld,
                          const _Float16* hsrc, _Float16* hdst, const float* bias,
                          _Float16* eout, int g)
{
  const int tid = threadIdx.x;
  // stage A: 32 rows x 512 halfs. x -> cols [0,256), h -> cols [256,512).
  // 2048 16-byte chunks total: 1024 for x, 1024 for h (4+4 per thread).
#if USE_ASYNC_LDS
  for (int i = tid; i < 1024; i += NTHR) {
    int r = i >> 5, cc = i & 31;
    async_b128(abuf + r * KC + cc * 8, xsrc + (size_t)r * xld + cc * 8);
  }
  for (int i = tid; i < 1024; i += NTHR) {
    int r = i >> 5, cc = i & 31;
    async_b128(abuf + r * KC + 256 + cc * 8, hsrc + r * HH + cc * 8);
  }
  wait_async0();
#else
  {
    v8h xv[4], hv[4];
#pragma unroll
    for (int j = 0; j < 4; ++j) {           // 8 global loads in flight
      int i = tid + j * NTHR;
      int r = i >> 5, cc = i & 31;
      xv[j] = *(const v8h*)(xsrc + (size_t)r * xld + cc * 8);
      hv[j] = *(const v8h*)(hsrc + r * HH + cc * 8);
    }
#pragma unroll
    for (int j = 0; j < 4; ++j) {           // then 8 LDS stores
      int i = tid + j * NTHR;
      int r = i >> 5, cc = i & 31;
      *(v8h*)(abuf + r * KC + cc * 8)       = xv[j];
      *(v8h*)(abuf + r * KC + 256 + cc * 8) = hv[j];
    }
  }
#endif
  __syncthreads();
  const int w = tid >> 5, lane = tid & 31;
  const int mt = w & 1, gate = w >> 1;   // 8 waves: 2 M-tiles x 4 gates
  v8f acc = zero8f();
  const _Float16* wbase = wlds + (size_t)(lw * 64 + gate * 16) * KC;
  const _Float16* atile = abuf + mt * 16 * KC;
#pragma unroll
  for (int kk = 0; kk < 16; ++kk) {
    v16h a = load_a(atile + kk * 32, KC, lane);
    v16h b = load_b(wbase + kk * 32, KC, lane);
    acc = __builtin_amdgcn_wmma_f32_16x16x32_f16(false, a, false, b, (short)0, acc, false, false);
  }
  {
    int n = lane & 15, mb = (lane < 16) ? 0 : 8;
#pragma unroll
    for (int r = 0; r < 8; ++r)
      gbufS[gate * 512 + (mt * 16 + mb + r) * 16 + n] = acc[r];
  }
  __syncthreads();
  // gate epilogue: 512 elements (32 batch x 16 local hidden)
  for (int e = tid; e < 512; e += NTHR) {
    int m = e >> 4, jl = e & 15;
    float iv = gbufS[e]        + bias[       16 * g + jl];
    float fv = gbufS[512 + e]  + bias[256 +  16 * g + jl];
    float gv = gbufS[1024 + e] + bias[512 +  16 * g + jl];
    float ov = gbufS[1536 + e] + bias[768 +  16 * g + jl];
    float cn = sigf(fv) * c_arr[e] + sigf(iv) * tanhf(gv);
    float hn = sigf(ov) * tanhf(cn);
    c_arr[e] = cn;
    _Float16 hv = (_Float16)hn;
    hdst[m * HH + 16 * g + jl] = hv;
    if (eout) eout[(size_t)m * SS * HH + 16 * g + jl] = hv;
  }
  // caller issues gbar()
}

__device__ void encode(const Ctx& c, const _Float16* wlds, _Float16* abuf, float* gbufS,
                       float* c_enc, _Float16* const h0g[2], _Float16* const h1g[2],
                       unsigned& lgen, int g)
{
  const int tid = threadIdx.x;
  // embed: emb[b,s,h] = tanh(x . W_emb[h,:] + b_emb[h]); x[7] comes from flag
  for (int i = (int)blockIdx.x * NTHR + tid; i < BB * SS * HH; i += NWG * NTHR) {
    int hcol = i & (HH - 1);
    int bs = i >> 8;
    float acc = c.b_emb[hcol];
    const float* wrow = c.W_emb + hcol * DD;
#pragma unroll
    for (int d = 0; d < DD - 1; ++d) acc += c.inputs[bs * DD + d] * wrow[d];
    acc += c.flag[bs] * wrow[DD - 1];
    c.emb[i] = (_Float16)tanhf(acc);
  }
  // zero recurrent state (fresh encode every call, as in reference)
  for (int i = tid; i < 1024; i += NTHR) c_enc[i] = 0.0f;
  v8h z = zero8h();
  for (int i = tid; i < 128; i += NTHR) {
    int buf = i >> 6, r = (i >> 1) & 31, cc = i & 1;
    _Float16* d = (buf ? h1g[0] : h0g[0]) + r * HH + 16 * g + cc * 8;
    *(v8h*)d = z;
  }
  gbar(c.bar, lgen);
  for (int t = 0; t < SS; ++t) {
    int p = t & 1, q = p ^ 1;
    lstm_cell(c, wlds, abuf, gbufS, c_enc,       0, c.emb + t * HH, SS * HH,
              h0g[p], h0g[q], c.bias0, nullptr, g);
    gbar(c.bar, lgen);
    lstm_cell(c, wlds, abuf, gbufS, c_enc + 512, 1, h0g[q],        HH,
              h1g[p], h1g[q], c.bias1, c.enc_out + t * HH, g);
    gbar(c.bar, lgen);
  }
}

// Attention + softmax + argmax + state updates for this WGP's 2 batch rows.
__device__ void attention(const Ctx& c, float* u2, float* scorePart, float* scoreL,
                          float* redL, const _Float16* dec_out, int step, int g)
{
  const int tid = threadIdx.x;
  const int w = tid >> 5, lane = tid & 31;
  // u2 = dec_out @ W2^T (32x256), computed redundantly per WGP (cheap)
  for (int tI = w; tI < 32; tI += 8) {
    int mt = tI & 1, nt = tI >> 1;
    v8f acc = zero8f();
#pragma unroll
    for (int kk = 0; kk < 8; ++kk) {
      v16h a = load_a(dec_out + mt * 16 * HH + kk * 32, HH, lane);
      v16h b = load_b(c.W2h + nt * 16 * HH + kk * 32, HH, lane);
      acc = __builtin_amdgcn_wmma_f32_16x16x32_f16(false, a, false, b, (short)0, acc, false, false);
    }
    int n = lane & 15, mb = (lane < 16) ? 0 : 8;
#pragma unroll
    for (int r = 0; r < 8; ++r)
      u2[(mt * 16 + mb + r) * HH + nt * 16 + n] = acc[r];
  }
  for (int i = tid; i < 8 * 128; i += NTHR) scorePart[i] = 0.0f;
  __syncthreads();
  // u1 rows for b in {2g, 2g+1}: 128 flattened rows of enc_out, fused tanh+dot(v)
  const _Float16* erows = c.enc_out + (size_t)(2 * g) * SS * HH;
  for (int tI = w; tI < 128; tI += 8) {
    int mt = tI >> 4, nt = tI & 15;
    v8f acc = zero8f();
#pragma unroll
    for (int kk = 0; kk < 8; ++kk) {
      v16h a = load_a(erows + mt * 16 * HH + kk * 32, HH, lane);
      v16h b = load_b(c.W1h + nt * 16 * HH + kk * 32, HH, lane);
      acc = __builtin_amdgcn_wmma_f32_16x16x32_f16(false, a, false, b, (short)0, acc, false, false);
    }
    int n = lane & 15, mb = (lane < 16) ? 0 : 8;
    int ncol = nt * 16 + n;
    float vn = c.vvec[ncol];
#pragma unroll
    for (int r = 0; r < 8; ++r) {
      int ml = mt * 16 + mb + r;              // local flattened row (2 batches x 64)
      int bg = 2 * g + (ml >> 6);
      float val = tanhf(acc[r] + u2[bg * HH + ncol]) * vn;
      val += __shfl_xor(val, 8, 16);
      val += __shfl_xor(val, 4, 16);
      val += __shfl_xor(val, 2, 16);
      val += __shfl_xor(val, 1, 16);
      if (n == 0) scorePart[w * 128 + ml] += val;   // deterministic per-wave partials
    }
  }
  __syncthreads();
  if (tid < 128) {
    float s = 0.0f;
#pragma unroll
    for (int ww = 0; ww < 8; ++ww) s += scorePart[ww * 128 + tid];
    int bg = 2 * g + (tid >> 6);
    if (c.flag[bg * SS + (tid & 63)] != 0.0f) s = -__builtin_huge_valf();
    scoreL[tid] = s;
  }
  __syncthreads();
  if (tid < 2) {   // per-batch max / argmax (first occurrence) / sum-exp
    int base = tid * 64;
    float mx = scoreL[base];
    int ix = 0;
    for (int s = 1; s < 64; ++s)
      if (scoreL[base + s] > mx) { mx = scoreL[base + s]; ix = s; }
    float sum = 0.0f;
    for (int s = 0; s < 64; ++s) sum += __expf(scoreL[base + s] - mx);
    redL[tid] = mx; redL[2 + tid] = sum; redL[4 + tid] = (float)ix;
    int bg = 2 * g + tid;
    c.out_idx[bg * SS + step] = (float)ix;
    c.flag[bg * SS + ix] = 1.0f;
  }
  __syncthreads();
  if (tid < 128) {
    int bl = tid >> 6, s = tid & 63, bg = 2 * g + bl;
    float pv = __expf(scoreL[tid] - redL[bl]) / redL[2 + bl];
    c.out_ptr[((size_t)bg * SS + step) * SS + s] = pv;
  }
  if (tid < 64) {  // dec_in[b] = enc_out[b, argmax, :]
    int bl = tid >> 5, cc = tid & 31, bg = 2 * g + bl;
    int ix = (int)redL[4 + bl];
    *(v8h*)(c.dec_in + bg * HH + cc * 8) =
        *(const v8h*)(c.enc_out + ((size_t)(bg * SS + ix)) * HH + cc * 8);
  }
  // caller issues gbar()
}

__global__ void __launch_bounds__(NTHR) ptrnet_main(Ctx c)
{
  extern __shared__ char smem[];
  _Float16* wlds = (_Float16*)smem;                                   // 4*64*512 fp16 = 256KB
  _Float16* abuf = (_Float16*)(smem + (size_t)4 * 64 * KC * sizeof(_Float16)); // 32x512 fp16 = 32KB
  float* u2 = (float*)abuf;                                           // union: 32x256 f32 = 32KB
  __shared__ float gbufS[4 * 32 * 16];
  __shared__ float c_enc[2 * 32 * 16];
  __shared__ float c_dec[2 * 32 * 16];
  __shared__ float scorePart[8 * 128];
  __shared__ float scoreL[128];
  __shared__ float redL[8];

  const int tid = threadIdx.x;
  const int g = blockIdx.x;        // this WGP owns hidden units [16g, 16g+16)
  unsigned lgen = 0;

  _Float16* h0g[2] = { c.h0g0, c.h0g1 };
  _Float16* h1g[2] = { c.h1g0, c.h1g1 };
  _Float16* hd0[2] = { c.hd00, c.hd01 };
  _Float16* hd1[2] = { c.hd10, c.hd11 };

  // Preload this WGP's slice of all 4 LSTM weight matrices into LDS (once).
  // Local row r (0..63): gate = r>>4, local j = r&15 -> global n = 256*gate + 16g + j.
#if USE_ASYNC_LDS
  for (int i = tid; i < 4 * 64 * 64; i += NTHR) {
    int l = i >> 12, rem = i & 4095;
    int r = rem >> 6, kc = rem & 63;
    int n = (r >> 4) * 256 + 16 * g + (r & 15);
    async_b128(wlds + (size_t)(l * 64 + r) * KC + kc * 8,
               c.Wp + (size_t)(l * G4 + n) * KC + kc * 8);
  }
  wait_async0();
#else
  for (int i = tid; i < 4 * 64 * 64; i += NTHR) {
    int l = i >> 12, rem = i & 4095;
    int r = rem >> 6, kc = rem & 63;
    int n = (r >> 4) * 256 + 16 * g + (r & 15);
    *(v8h*)(wlds + (size_t)(l * 64 + r) * KC + kc * 8) =
        *(const v8h*)(c.Wp + (size_t)(l * G4 + n) * KC + kc * 8);
  }
#endif
  __syncthreads();

  // ---- initial encode (also produces dec h/c init and dec_in0) ----
  encode(c, wlds, abuf, gbufS, c_enc, h0g, h1g, lgen, g);
  // final encoder h lives in parity-0 buffers (t=63 -> q=0)
  for (int i = tid; i < 128; i += NTHR) {
    int buf = i >> 6, r = (i >> 1) & 31, cc = i & 1;
    const _Float16* s = (buf ? h1g[0] : h0g[0]) + r * HH + 16 * g + cc * 8;
    _Float16* d = (buf ? hd1[0] : hd0[0]) + r * HH + 16 * g + cc * 8;
    *(v8h*)d = *(const v8h*)s;
  }
  for (int i = tid; i < 1024; i += NTHR) c_dec[i] = c_enc[i];
  for (int i = tid; i < 1024; i += NTHR) {   // dec_in = enc_out[:, 0, :]
    int b = i >> 5, cc = i & 31;
    *(v8h*)(c.dec_in + b * HH + cc * 8) =
        *(const v8h*)(c.enc_out + (size_t)b * SS * HH + cc * 8);
  }
  gbar(c.bar, lgen);

  // ---- decode loop ----
  for (int step = 0; step < SS; ++step) {
    encode(c, wlds, abuf, gbufS, c_enc, h0g, h1g, lgen, g);   // re-encode with updated flags
    int p = step & 1, q = p ^ 1;
    lstm_cell(c, wlds, abuf, gbufS, c_dec,       2, c.dec_in, HH,
              hd0[p], hd0[q], c.bias2, nullptr, g);
    gbar(c.bar, lgen);
    lstm_cell(c, wlds, abuf, gbufS, c_dec + 512, 3, hd0[q],   HH,
              hd1[p], hd1[q], c.bias3, nullptr, g);
    gbar(c.bar, lgen);
    attention(c, u2, scorePart, scoreL, redL, hd1[q], step, g);
    gbar(c.bar, lgen);
  }
}

// ---------- prep kernels ----------
__global__ void k_pack_lstm(const float* Wih, const float* Whh, _Float16* dst)
{
  int i = blockIdx.x * blockDim.x + threadIdx.x;
  if (i >= G4 * KC) return;
  int n = i >> 9, k = i & (KC - 1);
  float v = (k < HH) ? Wih[n * HH + k] : Whh[n * HH + (k - HH)];
  dst[i] = (_Float16)v;
}
__global__ void k_pack_f16(const float* src, _Float16* dst, int count)
{
  int i = blockIdx.x * blockDim.x + threadIdx.x;
  if (i < count) dst[i] = (_Float16)src[i];
}
__global__ void k_init(float* flag, unsigned* bar)
{
  int i = blockIdx.x * blockDim.x + threadIdx.x;
  if (i < BB * SS) flag[i] = 0.0f;
  if (i < 2) bar[i] = 0u;
}

extern "C" void kernel_launch(void* const* d_in, const int* in_sizes, int n_in,
                              void* d_out, int out_size, void* d_ws, size_t ws_size,
                              hipStream_t stream)
{
  (void)in_sizes; (void)n_in; (void)out_size;
  char* ws = (char*)d_ws;
  size_t off = 0;
  auto take = [&](size_t bytes) -> char* {
    char* p = ws + off;
    off += (bytes + 255) & ~(size_t)255;
    return p;
  };
  _Float16* Wp      = (_Float16*)take((size_t)4 * G4 * KC * sizeof(_Float16));
  _Float16* W1h     = (_Float16*)take((size_t)HH * HH * sizeof(_Float16));
  _Float16* W2h     = (_Float16*)take((size_t)HH * HH * sizeof(_Float16));
  _Float16* emb     = (_Float16*)take((size_t)BB * SS * HH * sizeof(_Float16));
  _Float16* enc_out = (_Float16*)take((size_t)BB * SS * HH * sizeof(_Float16));
  _Float16* hbufs   = (_Float16*)take((size_t)8 * BB * HH * sizeof(_Float16));
  _Float16* dec_in  = (_Float16*)take((size_t)BB * HH * sizeof(_Float16));
  float*    flag    = (float*)take((size_t)BB * SS * sizeof(float));
  unsigned* bar     = (unsigned*)take(256);
  if (off > ws_size) return;

  const int PK = G4 * KC;
  k_pack_lstm<<<(PK + 255) / 256, 256, 0, stream>>>((const float*)d_in[3],  (const float*)d_in[4],  Wp + 0 * (size_t)PK);
  k_pack_lstm<<<(PK + 255) / 256, 256, 0, stream>>>((const float*)d_in[6],  (const float*)d_in[7],  Wp + 1 * (size_t)PK);
  k_pack_lstm<<<(PK + 255) / 256, 256, 0, stream>>>((const float*)d_in[9],  (const float*)d_in[10], Wp + 2 * (size_t)PK);
  k_pack_lstm<<<(PK + 255) / 256, 256, 0, stream>>>((const float*)d_in[12], (const float*)d_in[13], Wp + 3 * (size_t)PK);
  k_pack_f16<<<(HH * HH + 255) / 256, 256, 0, stream>>>((const float*)d_in[15], W1h, HH * HH);
  k_pack_f16<<<(HH * HH + 255) / 256, 256, 0, stream>>>((const float*)d_in[16], W2h, HH * HH);
  k_init<<<(BB * SS + 255) / 256, 256, 0, stream>>>(flag, bar);

  Ctx c{};
  c.inputs = (const float*)d_in[0];
  c.W_emb  = (const float*)d_in[1];
  c.b_emb  = (const float*)d_in[2];
  c.bias0  = (const float*)d_in[5];
  c.bias1  = (const float*)d_in[8];
  c.bias2  = (const float*)d_in[11];
  c.bias3  = (const float*)d_in[14];
  c.vvec   = (const float*)d_in[17];
  c.Wp = Wp; c.W1h = W1h; c.W2h = W2h;
  c.emb = emb; c.enc_out = enc_out;
  c.h0g0 = hbufs + 0 * (BB * HH); c.h0g1 = hbufs + 1 * (BB * HH);
  c.h1g0 = hbufs + 2 * (BB * HH); c.h1g1 = hbufs + 3 * (BB * HH);
  c.hd00 = hbufs + 4 * (BB * HH); c.hd01 = hbufs + 5 * (BB * HH);
  c.hd10 = hbufs + 6 * (BB * HH); c.hd11 = hbufs + 7 * (BB * HH);
  c.dec_in = dec_in; c.flag = flag; c.bar = bar;
  c.out_ptr = (float*)d_out;
  c.out_idx = (float*)d_out + (size_t)BB * SS * SS;

  const size_t smem = (size_t)(4 * 64 * KC + BB * KC) * sizeof(_Float16); // 288 KB dynamic LDS
  (void)hipFuncSetAttribute((const void*)ptrnet_main,
                            hipFuncAttributeMaxDynamicSharedMemorySize, (int)smem);
  ptrnet_main<<<NWG, NTHR, smem, stream>>>(c);
}
